// RNNDecoder_18073222381757
// MI455X (gfx1250) — matrix operand order repaired and checked by reading, not compiled, not run on hardware
//
#include <hip/hip_runtime.h>
#include <stdint.h>

// Problem constants (match reference)
#define B_   8
#define S_   256
#define V_   32000
#define E_   512
#define H_   1024
#define G3H_ (3 * H_)   // 3072
#define BS_  (B_ * S_)  // 2048

typedef __attribute__((ext_vector_type(16))) __bf16 v16bf;
typedef __attribute__((ext_vector_type(8)))  float  v8f;

union TileBF { v16bf v; uint4 q[2]; };

__device__ __forceinline__ unsigned short f2bf(float f) {
  unsigned int u = __float_as_uint(f);
  u += 0x7FFFu + ((u >> 16) & 1u);   // round-to-nearest-even
  return (unsigned short)(u >> 16);
}

// A tile (16xK32, 16-bit): lane<16 -> M=lane, K {k0..k0+7, k0+16..k0+23}
//                          lane>=16 -> M=lane-16, K {k0+8..k0+15, k0+24..k0+31}
__device__ __forceinline__ v16bf load_a16(const unsigned short* __restrict__ A,
                                          int lda, int m0, int k0, int lane) {
  const unsigned short* p =
      A + (size_t)(m0 + (lane & 15)) * lda + (size_t)k0 + ((lane < 16) ? 0 : 8);
  TileBF t;
  t.q[0] = *(const uint4*)(p);
  t.q[1] = *(const uint4*)(p + 16);
  return t.v;
}

// B tile (K32x16) from a row-major [N,K] matrix (multiply by its transpose):
// lane<16 -> N=lane, K {k0..k0+7, k0+16..k0+23}; lane>=16 -> N=lane-16, K {k0+8..k0+15, k0+24..k0+31}
__device__ __forceinline__ v16bf load_b16(const unsigned short* __restrict__ Bm,
                                          int ldb, int n0, int k0, int lane) {
  const unsigned short* p =
      Bm + (size_t)(n0 + (lane & 15)) * ldb + (size_t)k0 + ((lane < 16) ? 0 : 8);
  TileBF t;
  t.q[0] = *(const uint4*)(p);
  t.q[1] = *(const uint4*)(p + 16);
  return t.v;
}

__device__ __forceinline__ v8f wmma_bf16(v16bf a, v16bf b, v8f c) {
  // 8 args: (neg_a, A, neg_b, B, c_mod, C, reuse_a, reuse_b)
  return __builtin_amdgcn_wmma_f32_16x16x32_bf16(false, a, false, b, (short)0, c,
                                                 false, false);
}

__device__ __forceinline__ float sigmoidf_(float x) {
  return 1.0f / (1.0f + __expf(-x));
}

// ---------------------------------------------------------------- utilities
__global__ void __launch_bounds__(256) k_cvt_bf16(const float* __restrict__ src,
                                                  unsigned short* __restrict__ dst,
                                                  int n) {
  int i = blockIdx.x * blockDim.x + threadIdx.x;
  int st = gridDim.x * blockDim.x;
  for (; i < n; i += st) dst[i] = f2bf(src[i]);
}

__global__ void __launch_bounds__(256) k_zero(unsigned int* __restrict__ p, int nwords) {
  int i = blockIdx.x * blockDim.x + threadIdx.x;
  int st = gridDim.x * blockDim.x;
  for (; i < nwords; i += st) p[i] = 0u;
}

// gather: emb_bf[row, :] = bf16(emb_W[x[row], :])
__global__ void __launch_bounds__(128) k_embed(const int* __restrict__ x,
                                               const float* __restrict__ embW,
                                               unsigned short* __restrict__ emb_bf) {
  int row = blockIdx.x;                 // 0..2047
  int tok = x[row];
  const float* src = embW + (size_t)tok * E_;
  for (int j = threadIdx.x; j < E_; j += blockDim.x)
    emb_bf[(size_t)row * E_ + j] = f2bf(src[j]);
}

// ------------------------------------------------- gx = emb @ W_ih^T + b_ih
// 4 M-tiles per wave: B tile reused 4x, 4 independent accumulators.
__global__ void __launch_bounds__(256)
k_gemm_gx(const unsigned short* __restrict__ Abf,  // emb_bf [2048,512]
          const unsigned short* __restrict__ Wbf,  // W_ih_bf [3072,512]
          const float* __restrict__ bias,          // b_ih [3072]
          float* __restrict__ C) {                 // gx [2048,3072]
  const int NT = G3H_ / 16;                        // 192
  int tile = blockIdx.x * 8 + (threadIdx.x >> 5);  // 0..6143
  int lane = threadIdx.x & 31;
  int m0 = (tile / NT) * 64;
  int n0 = (tile % NT) * 16;
  v8f acc0 = {0.f,0.f,0.f,0.f,0.f,0.f,0.f,0.f};
  v8f acc1 = acc0, acc2 = acc0, acc3 = acc0;
  for (int k0 = 0; k0 < E_; k0 += 32) {
    v16bf b  = load_b16(Wbf, E_, n0, k0, lane);
    v16bf a0 = load_a16(Abf, E_, m0 +  0, k0, lane);
    v16bf a1 = load_a16(Abf, E_, m0 + 16, k0, lane);
    v16bf a2 = load_a16(Abf, E_, m0 + 32, k0, lane);
    v16bf a3 = load_a16(Abf, E_, m0 + 48, k0, lane);
    acc0 = wmma_bf16(a0, b, acc0);
    acc1 = wmma_bf16(a1, b, acc1);
    acc2 = wmma_bf16(a2, b, acc2);
    acc3 = wmma_bf16(a3, b, acc3);
  }
  int n = n0 + (lane & 15);
  float bj = bias[n];
  int mb = (lane < 16) ? 0 : 8;
#pragma unroll
  for (int v = 0; v < 8; ++v) {
    C[(size_t)(m0 +  0 + mb + v) * G3H_ + n] = acc0[v] + bj;
    C[(size_t)(m0 + 16 + mb + v) * G3H_ + n] = acc1[v] + bj;
    C[(size_t)(m0 + 32 + mb + v) * G3H_ + n] = acc2[v] + bj;
    C[(size_t)(m0 + 48 + mb + v) * G3H_ + n] = acc3[v] + bj;
  }
}

// ------------------------------------------------------------- GRU (persistent)
// 32 blocks x 192 threads (6 waves). Block b owns h-columns [b*32, b*32+32) and
// the matching r/z/n gate tiles; one device barrier per step; h double-buffered.
#define GRU_NBLK 32
__global__ void __launch_bounds__(192, 1)
k_gru(const int* __restrict__ x,             // [8,256]
      const float* __restrict__ gx,          // [2048,3072]
      const float* __restrict__ bhh,         // [3072]
      const unsigned short* __restrict__ Whh_bf,  // [3072,1024]
      float* __restrict__ h32,               // [2][8][1024]
      unsigned short* __restrict__ h16,      // [2][16][1024] rows 8..15 zero
      unsigned short* __restrict__ ctx_bf,   // [2048,1024]
      unsigned int* __restrict__ bar) {      // [256]
  __shared__ float lds_g[3][8][32];
  const int tid  = threadIdx.x;
  const int lane = tid & 31;
  const int w    = tid >> 5;                 // 0..5
  const int cg   = blockIdx.x * 2 + ((w >= 3) ? 1 : 0);  // column group 0..63
  const int gi   = w % 3;                    // 0=r 1=z 2=n
  const int n0   = gi * H_ + cg * 16;        // gate-column tile base in [0,3072)
  const int colb = blockIdx.x * 32;          // h-column base for this block

#pragma unroll 1
  for (int t = 0; t < S_; ++t) {
    const int rb = t & 1, wb = (t + 1) & 1;
    // ---- WMMA phase: g_h tile = h @ W_hh^T (M=16 pad, N=16, K=1024)
    const unsigned short* ap = h16 + (size_t)rb * (16 * H_) +
                               (size_t)(lane & 15) * H_ + ((lane < 16) ? 0 : 8);
    const unsigned short* bp = Whh_bf + (size_t)(n0 + (lane & 15)) * H_ +
                               ((lane < 16) ? 0 : 8);
    v8f acc = {0.f, 0.f, 0.f, 0.f, 0.f, 0.f, 0.f, 0.f};
#pragma unroll 4
    for (int kk = 0; kk < H_ / 32; ++kk) {
      TileBF a, b;
      a.q[0] = *(const uint4*)(ap);
      a.q[1] = *(const uint4*)(ap + 16);
      b.q[0] = *(const uint4*)(bp);
      b.q[1] = *(const uint4*)(bp + 16);
      acc = wmma_bf16(a.v, b.v, acc);
      ap += 32;
      bp += 32;
    }
    // batch rows 0..7 live in lanes 0..15, VGPRs 0..7
    if (lane < 16) {
      float bh = bhh[n0 + lane];
      int lc = ((w >= 3) ? 16 : 0) + lane;
#pragma unroll
      for (int v = 0; v < 8; ++v) lds_g[gi][v][lc] = acc[v] + bh;
    }
    __syncthreads();
    // ---- gate nonlinearity + h update for this block's 8x32 slice
    const float* h32o = h32 + (size_t)rb * (8 * H_);
    float*       h32n = h32 + (size_t)wb * (8 * H_);
    unsigned short* h16n = h16 + (size_t)wb * (16 * H_);
    for (int idx = tid; idx < 256; idx += 192) {
      int m = idx >> 5, c = idx & 31;
      int i = colb + c;
      size_t grow = ((size_t)m * S_ + t) * G3H_;
      float ghr = lds_g[0][m][c], ghz = lds_g[1][m][c], ghn = lds_g[2][m][c];
      float r  = sigmoidf_(gx[grow + i] + ghr);
      float z  = sigmoidf_(gx[grow + H_ + i] + ghz);
      float nn = tanhf(gx[grow + 2 * H_ + i] + r * ghn);
      float hold = h32o[m * H_ + i];
      float hnew = (1.0f - z) * nn + z * hold;
      bool pad = (x[m * S_ + t] == 0);          // PAD == 0
      float hc = pad ? hold : hnew;
      h32n[m * H_ + i] = hc;
      h16n[m * H_ + i] = f2bf(hc);
      ctx_bf[((size_t)m * S_ + t) * H_ + i] = pad ? (unsigned short)0 : f2bf(hc);
    }
    // ---- device-wide barrier (per-step counter slot, zeroed each launch)
    __builtin_amdgcn_fence(__ATOMIC_RELEASE, "agent");
    __syncthreads();
    if (tid == 0) {
      __hip_atomic_fetch_add(&bar[t], 1u, __ATOMIC_RELEASE, __HIP_MEMORY_SCOPE_AGENT);
      while (__hip_atomic_load(&bar[t], __ATOMIC_ACQUIRE, __HIP_MEMORY_SCOPE_AGENT) <
             (unsigned)GRU_NBLK) {
        __builtin_amdgcn_s_sleep(2);
      }
    }
    __syncthreads();
    __builtin_amdgcn_fence(__ATOMIC_ACQUIRE, "agent");
  }
}

// --------------------- logits = tanh([ctx, emb] @ lin_W^T + lin_b), K = 1024+512
__global__ void __launch_bounds__(256)
k_gemm_logits(const unsigned short* __restrict__ ctx_bf,  // [2048,1024]
              const unsigned short* __restrict__ emb_bf,  // [2048,512]
              const unsigned short* __restrict__ linW_bf, // [512,1536]
              const float* __restrict__ lin_b,            // [512]
              unsigned short* __restrict__ logits_bf) {   // [2048,512]
  const int NT = E_ / 16;  // 32
  int tile = blockIdx.x * 8 + (threadIdx.x >> 5);  // 0..1023
  int lane = threadIdx.x & 31;
  int m0 = (tile / NT) * 64;
  int n0 = (tile % NT) * 16;
  const int LDB = H_ + E_;  // 1536
  v8f acc0 = {0.f,0.f,0.f,0.f,0.f,0.f,0.f,0.f};
  v8f acc1 = acc0, acc2 = acc0, acc3 = acc0;
  for (int k0 = 0; k0 < H_; k0 += 32) {           // ctx part
    v16bf b  = load_b16(linW_bf, LDB, n0, k0, lane);
    v16bf a0 = load_a16(ctx_bf, H_, m0 +  0, k0, lane);
    v16bf a1 = load_a16(ctx_bf, H_, m0 + 16, k0, lane);
    v16bf a2 = load_a16(ctx_bf, H_, m0 + 32, k0, lane);
    v16bf a3 = load_a16(ctx_bf, H_, m0 + 48, k0, lane);
    acc0 = wmma_bf16(a0, b, acc0);
    acc1 = wmma_bf16(a1, b, acc1);
    acc2 = wmma_bf16(a2, b, acc2);
    acc3 = wmma_bf16(a3, b, acc3);
  }
  for (int k0 = 0; k0 < E_; k0 += 32) {           // emb part
    v16bf b  = load_b16(linW_bf, LDB, n0, H_ + k0, lane);
    v16bf a0 = load_a16(emb_bf, E_, m0 +  0, k0, lane);
    v16bf a1 = load_a16(emb_bf, E_, m0 + 16, k0, lane);
    v16bf a2 = load_a16(emb_bf, E_, m0 + 32, k0, lane);
    v16bf a3 = load_a16(emb_bf, E_, m0 + 48, k0, lane);
    acc0 = wmma_bf16(a0, b, acc0);
    acc1 = wmma_bf16(a1, b, acc1);
    acc2 = wmma_bf16(a2, b, acc2);
    acc3 = wmma_bf16(a3, b, acc3);
  }
  int n = n0 + (lane & 15);
  float bj = lin_b[n];
  int mb = (lane < 16) ? 0 : 8;
#pragma unroll
  for (int v = 0; v < 8; ++v) {
    logits_bf[(size_t)(m0 +  0 + mb + v) * E_ + n] = f2bf(tanhf(acc0[v] + bj));
    logits_bf[(size_t)(m0 + 16 + mb + v) * E_ + n] = f2bf(tanhf(acc1[v] + bj));
    logits_bf[(size_t)(m0 + 32 + mb + v) * E_ + n] = f2bf(tanhf(acc2[v] + bj));
    logits_bf[(size_t)(m0 + 48 + mb + v) * E_ + n] = f2bf(tanhf(acc3[v] + bj));
  }
}

// --------------------------------------- scores = logits @ emb_W^T  (the big one)
// 4 M-tiles per wave: each emb_W tile is read once per 64 output rows instead of
// once per 16 -> B-matrix L2 traffic drops from ~4.2 GB to ~1 GB, and the four
// independent accumulators let consecutive v_wmma issue without RAW stalls.
__global__ void __launch_bounds__(256)
k_gemm_scores(const unsigned short* __restrict__ logits_bf, // [2048,512]
              const unsigned short* __restrict__ embW_bf,   // [32000,512]
              float* __restrict__ out) {                    // [2048,32000]
  const int NT = V_ / 16;  // 2000
  int tile = blockIdx.x * 8 + (threadIdx.x >> 5);  // 0..63999
  int lane = threadIdx.x & 31;
  int m0 = (tile / NT) * 64;
  int n0 = (tile % NT) * 16;
  v8f acc0 = {0.f,0.f,0.f,0.f,0.f,0.f,0.f,0.f};
  v8f acc1 = acc0, acc2 = acc0, acc3 = acc0;
  const unsigned short* brow = embW_bf + (size_t)(n0 + (lane & 15)) * E_;
  for (int k0 = 0; k0 < E_; k0 += 32) {
    if (k0 + 64 < E_) __builtin_prefetch(brow + k0 + 64, 0, 1);  // global_prefetch_b8
    v16bf b  = load_b16(embW_bf, E_, n0, k0, lane);
    v16bf a0 = load_a16(logits_bf, E_, m0 +  0, k0, lane);
    v16bf a1 = load_a16(logits_bf, E_, m0 + 16, k0, lane);
    v16bf a2 = load_a16(logits_bf, E_, m0 + 32, k0, lane);
    v16bf a3 = load_a16(logits_bf, E_, m0 + 48, k0, lane);
    acc0 = wmma_bf16(a0, b, acc0);
    acc1 = wmma_bf16(a1, b, acc1);
    acc2 = wmma_bf16(a2, b, acc2);
    acc3 = wmma_bf16(a3, b, acc3);
  }
  int n = n0 + (lane & 15);
  int mb = (lane < 16) ? 0 : 8;
#pragma unroll
  for (int v = 0; v < 8; ++v) {
    out[(size_t)(m0 +  0 + mb + v) * V_ + n] = acc0[v];
    out[(size_t)(m0 + 16 + mb + v) * V_ + n] = acc1[v];
    out[(size_t)(m0 + 32 + mb + v) * V_ + n] = acc2[v];
    out[(size_t)(m0 + 48 + mb + v) * V_ + n] = acc3[v];
  }
}

// ------------------- in-place row log_softmax over V (online max+sum, one read)
__global__ void __launch_bounds__(256) k_logsoftmax(float* __restrict__ out) {
  __shared__ float red_m[256];
  __shared__ float red_s[256];
  const int row = blockIdx.x;
  float* p = out + (size_t)row * V_;
  // online logsumexp per thread
  float mx = -3.402823466e38f, sum = 0.f;
  for (int j = threadIdx.x; j < V_; j += 256) {
    float v = p[j];
    if (v > mx) {
      sum = sum * expf(mx - v) + 1.0f;
      mx = v;
    } else {
      sum += expf(v - mx);
    }
  }
  red_m[threadIdx.x] = mx;
  red_s[threadIdx.x] = sum;
  __syncthreads();
  for (int s = 128; s > 0; s >>= 1) {
    if (threadIdx.x < s) {
      float m1 = red_m[threadIdx.x], m2 = red_m[threadIdx.x + s];
      float s1 = red_s[threadIdx.x], s2 = red_s[threadIdx.x + s];
      float M = fmaxf(m1, m2);
      red_m[threadIdx.x] = M;
      red_s[threadIdx.x] = s1 * expf(m1 - M) + s2 * expf(m2 - M);
    }
    __syncthreads();
  }
  float lse = red_m[0] + logf(red_s[0]);
  for (int j = threadIdx.x; j < V_; j += 256) p[j] -= lse;
}

// =============================================================== host launcher
extern "C" void kernel_launch(void* const* d_in, const int* in_sizes, int n_in,
                              void* d_out, int out_size, void* d_ws, size_t ws_size,
                              hipStream_t stream) {
  const int*   x    = (const int*)d_in[0];
  const float* embW = (const float*)d_in[1];
  const float* Wih  = (const float*)d_in[2];
  const float* Whh  = (const float*)d_in[3];
  const float* bih  = (const float*)d_in[4];
  const float* bhh  = (const float*)d_in[5];
  const float* linW = (const float*)d_in[6];
  const float* linb = (const float*)d_in[7];
  float* out = (float*)d_out;
  char*  ws  = (char*)d_ws;

  // workspace layout (all sizes multiples of 256 -> naturally aligned)
  constexpr size_t SZ_WIH  = (size_t)G3H_ * E_ * 2;        // 3,145,728
  constexpr size_t SZ_WHH  = (size_t)G3H_ * H_ * 2;        // 6,291,456
  constexpr size_t SZ_LINW = (size_t)E_ * (H_ + E_) * 2;   // 1,572,864
  constexpr size_t SZ_EMBW = (size_t)V_ * E_ * 2;          // 32,768,000
  constexpr size_t SZ_EMB  = (size_t)BS_ * E_ * 2;         // 2,097,152
  constexpr size_t SZ_GX   = (size_t)BS_ * G3H_ * 4;       // 25,165,824
  constexpr size_t SZ_CTX  = (size_t)BS_ * H_ * 2;         // 4,194,304
  constexpr size_t SZ_LOG  = (size_t)BS_ * E_ * 2;         // 2,097,152
  constexpr size_t SZ_H32  = (size_t)2 * 8 * H_ * 4;       // 65,536
  constexpr size_t SZ_H16  = (size_t)2 * 16 * H_ * 2;      // 65,536
  constexpr size_t SZ_BAR  = 256 * 4;                      // 1,024

  constexpr size_t O_WIH  = 0;
  constexpr size_t O_WHH  = O_WIH + SZ_WIH;
  constexpr size_t O_LINW = O_WHH + SZ_WHH;
  constexpr size_t O_EMBW = O_LINW + SZ_LINW;
  constexpr size_t O_EMB  = O_EMBW + SZ_EMBW;
  constexpr size_t O_GX   = O_EMB + SZ_EMB;
  constexpr size_t O_CTX  = O_GX + SZ_GX;
  constexpr size_t O_LOG  = O_CTX + SZ_CTX;
  constexpr size_t O_H32  = O_LOG + SZ_LOG;
  constexpr size_t O_H16  = O_H32 + SZ_H32;
  constexpr size_t O_BAR  = O_H16 + SZ_H16;

  unsigned short* wih_bf  = (unsigned short*)(ws + O_WIH);
  unsigned short* whh_bf  = (unsigned short*)(ws + O_WHH);
  unsigned short* linW_bf = (unsigned short*)(ws + O_LINW);
  unsigned short* embW_bf = (unsigned short*)(ws + O_EMBW);
  unsigned short* emb_bf  = (unsigned short*)(ws + O_EMB);
  float*          gx      = (float*)(ws + O_GX);
  unsigned short* ctx_bf  = (unsigned short*)(ws + O_CTX);
  unsigned short* logits_bf = (unsigned short*)(ws + O_LOG);
  float*          h32     = (float*)(ws + O_H32);
  unsigned short* h16     = (unsigned short*)(ws + O_H16);
  unsigned int*   bar     = (unsigned int*)(ws + O_BAR);

  // 1) weight conversions to bf16
  k_cvt_bf16<<<1024, 256, 0, stream>>>(Wih,  wih_bf,  G3H_ * E_);
  k_cvt_bf16<<<1024, 256, 0, stream>>>(Whh,  whh_bf,  G3H_ * H_);
  k_cvt_bf16<<<512,  256, 0, stream>>>(linW, linW_bf, E_ * (H_ + E_));
  k_cvt_bf16<<<2048, 256, 0, stream>>>(embW, embW_bf, V_ * E_);
  // 2) zero h (both buffers, incl. pad rows 8..15) + per-step barrier counters
  k_zero<<<132, 256, 0, stream>>>((unsigned int*)(ws + O_H32),
                                  (int)((SZ_H32 + SZ_H16 + SZ_BAR) / 4));
  // 3) embedding gather (bf16)
  k_embed<<<BS_, 128, 0, stream>>>(x, embW, emb_bf);
  // 4) gx = emb @ W_ih^T + b_ih     (32 Mgroups * 192 Ntiles / 8 waves)
  k_gemm_gx<<<(32 * 192) / 8, 256, 0, stream>>>(emb_bf, wih_bf, bih, gx);
  // 5) GRU recurrence (persistent, WMMA inner GEMM, 1 device barrier/step)
  k_gru<<<GRU_NBLK, 192, 0, stream>>>(x, gx, bhh, whh_bf, h32, h16, ctx_bf, bar);
  // 6) logits = tanh([ctx,emb] @ lin_W^T + b)   (32 * 32 / 8 waves)
  k_gemm_logits<<<(32 * 32) / 8, 256, 0, stream>>>(ctx_bf, emb_bf, linW_bf, linb,
                                                   logits_bf);
  // 7) scores = logits @ emb_W^T   (32 Mgroups * 2000 Ntiles / 8 waves)
  k_gemm_scores<<<(32 * 2000) / 8, 256, 0, stream>>>(logits_bf, embW_bf, out);
  // 8) in-place online log_softmax per row
  k_logsoftmax<<<BS_, 256, 0, stream>>>(out);

  (void)in_sizes; (void)n_in; (void)out_size; (void)ws_size;
}
